// SwitchTransformerEncoderLayer_31739808317982
// MI455X (gfx1250) — matrix operand order repaired and checked by reading, not compiled, not run on hardware
//
#include <hip/hip_runtime.h>
#include <math.h>
#include <stdint.h>

// ---------------------------------------------------------------------------
// Switch-Transformer encoder layer for gfx1250 (MI455X), wave32 + WMMA bf16,
// async global->LDS staging (ASYNCcnt) for GEMM B-tiles and attention K/V,
// register-pipelined A fragments, batched B-fragment LDS reads so WMMAs
// overlap DS latency (partial s_wait_dscnt instead of 0-waits).
// ---------------------------------------------------------------------------

#define DEV static __device__ __forceinline__

typedef __attribute__((ext_vector_type(16))) __bf16 v16bf;
typedef __attribute__((ext_vector_type(8)))  float  v8f;

constexpr int Bc = 2, Sc = 2048, Dc = 1024, Hc = 16, FFc = 4096, Ec = 8;
constexpr int HDc = Dc / Hc;     // 64
constexpr int Tc  = Bc * Sc;     // 4096
constexpr int D3c = 3 * Dc;      // 3072

union AF { v16bf v; __bf16 e[16]; };
union CF { v8f   v; float  e[8];  };

DEV void zeroC(CF& c) {
#pragma unroll
  for (int i = 0; i < 8; i++) c.e[i] = 0.f;
}

// A fragment 16x32 (MxK), per-lane row pointer (for gathered rows).
// lane L: m=L&15, hi=L>>4. elements 0..15 = row[k0+hi*16 .. +15]
DEV void loadA_rp(AF& f, const __bf16* rowp, int k0, int lane) {
  int hi = lane >> 4;
  const __bf16* p = rowp + k0 + hi * 16;
#pragma unroll
  for (int i = 0; i < 16; i++) f.e[i] = p[i];
}

DEV void loadA(AF& f, const __bf16* src, int ld, int k0, int lane) {
  int m = lane & 15;
  loadA_rp(f, src + (size_t)m * ld, k0, lane);
}

// B fragment 32x16 (KxN) from row-major [N x K] source (ld = K stride).
// lane L: n=L&15, kb=(L>>4)*16; 16 contiguous K values of row n.
DEV void loadB_rm(AF& f, const __bf16* src, int ld, int k0, int lane) {
  int n = lane & 15, kb = (lane >> 4) * 16;
  const __bf16* p = src + (size_t)n * ld + k0 + kb;
#pragma unroll
  for (int i = 0; i < 16; i++) f.e[i] = p[i];
}

// B fragment 32x16 (KxN) from row-major [K x ld] source (e.g. V[s,hd]).
DEV void loadB_cm(AF& f, const __bf16* src, int ld, int k0, int lane) {
  int n = lane & 15, kb = (lane >> 4) * 16;
  const __bf16* p = src + (size_t)(k0 + kb) * ld + n;
#pragma unroll
  for (int i = 0; i < 16; i++) f.e[i] = p[(size_t)i * ld];
}

DEV void mma(CF& c, const AF& a, const AF& b) {
  c.v = __builtin_amdgcn_wmma_f32_16x16x32_bf16(false, a.v, false, b.v,
                                                (short)0, c.v, false, false);
}

// --- async global->LDS (CDNA5, tracked by ASYNCcnt) ------------------------
DEV void async_b128(uint32_t ldsoff, const __bf16* gaddr) {
  asm volatile("global_load_async_to_lds_b128 %0, %1, off"
               :: "v"(ldsoff), "v"((uint64_t)(uintptr_t)gaddr)
               : "memory");
}
#define WAITA(n) asm volatile("s_wait_asynccnt " #n ::: "memory")

DEV uint32_t lds_off(const void* p) { return (uint32_t)(uintptr_t)p; }

// Stage one 64x32 bf16 B tile (rows n0.., 32 K) into LDS: 256 chunks of 16B,
// 2 async ops per thread (= 2 per wave per stage).
DEV void issue_b(const __bf16* bsrc, size_t ldk, int k0, uint32_t lbase, int tid) {
#pragma unroll
  for (int h = 0; h < 2; h++) {
    int c = tid + h * 128;
    int row = c >> 2, part = c & 3;
    async_b128(lbase + row * 64 + part * 16,
               bsrc + (size_t)row * ldk + k0 + part * 8);
  }
}

// Stage one 64x64 bf16 tile (row-major, ld=64): 512 chunks of 16B.
DEV void issue_tile64(const __bf16* src, uint32_t lbase, int tid) {
#pragma unroll
  for (int h = 0; h < 4; h++) {
    int c = tid + h * 128;
    int row = c >> 3, part = c & 7;
    async_b128(lbase + row * 128 + part * 16,
               src + (size_t)row * 64 + part * 8);
  }
}

// Double-buffered K-loop: acc[4] += A(16 x 32*nk) @ B(64 x 32*nk)^T.
// B tiles double-buffered in LDS via async copies; A fragments pipelined in
// registers; all 4 B fragments loaded before the WMMA burst so the DS reads
// clause up and overlap with the matrix pipe.
DEV void gemm_kloop(CF acc[4], const __bf16* arow, const __bf16* bsrc, size_t ldk,
                    int nk, __bf16* btile, int tid, int lane) {
  uint32_t lb0 = lds_off(btile), lb1 = lds_off(btile + 64 * 32);
  issue_b(bsrc, ldk, 0, lb0, tid);
  AF a_cur;
  loadA_rp(a_cur, arow, 0, lane);
  for (int i = 0; i < nk; i++) {
    if (i + 1 < nk) {
      issue_b(bsrc, ldk, (i + 1) * 32, ((i + 1) & 1) ? lb1 : lb0, tid);
      WAITA(2);          // previous stage done (per-wave, in-order)
    } else {
      WAITA(0);
    }
    __syncthreads();     // all waves' chunks visible
    AF a_next;
    if (i + 1 < nk) {
      loadA_rp(a_next, arow, (i + 1) * 32, lane);            // hide behind WMMAs
      if (i + 4 < nk) __builtin_prefetch(arow + (i + 4) * 32, 0, 3);
    }
    const __bf16* tp = btile + (i & 1) * (64 * 32);
    AF b0, b1, b2, b3;                 // batched DS reads -> partial dscnt waits
    loadB_rm(b0, tp + 0 * 16 * 32, 32, 0, lane);
    loadB_rm(b1, tp + 1 * 16 * 32, 32, 0, lane);
    loadB_rm(b2, tp + 2 * 16 * 32, 32, 0, lane);
    loadB_rm(b3, tp + 3 * 16 * 32, 32, 0, lane);
    mma(acc[0], a_cur, b0);
    mma(acc[1], a_cur, b1);
    mma(acc[2], a_cur, b2);
    mma(acc[3], a_cur, b3);
    __syncthreads();     // done reading before buffer is reused
    if (i + 1 < nk) a_cur = a_next;
  }
}

// ---------------------------------------------------------------------------
__global__ void k_cvt_bf16(const float* __restrict__ in, __bf16* __restrict__ out,
                           size_t n) {
  size_t i  = (size_t)blockIdx.x * blockDim.x + threadIdx.x;
  size_t st = (size_t)gridDim.x * blockDim.x;
  for (; i < n; i += st) out[i] = (__bf16)in[i];
}

// ---------------------------------------------------------------------------
// QKV: C[t,n] = x @ wqkv^T + b ; scatter to q/k/v [B,H,S,HD] bf16, q scaled.
__global__ void __launch_bounds__(128)
k_qkv(const __bf16* __restrict__ xb, const __bf16* __restrict__ wq,
      const float* __restrict__ bq, __bf16* __restrict__ qb,
      __bf16* __restrict__ kb, __bf16* __restrict__ vb) {
  __shared__ __bf16 btile[2 * 64 * 32];
  int tid = threadIdx.x, lane = tid & 31, w = tid >> 5;
  int m0 = blockIdx.y * 64 + w * 16;
  int n0 = blockIdx.x * 64;
  int col = lane & 15, hi = lane >> 4;
  CF acc[4];
#pragma unroll
  for (int i = 0; i < 4; i++) zeroC(acc[i]);
  const __bf16* arow = xb + (size_t)(m0 + col) * Dc;
  gemm_kloop(acc, arow, wq + (size_t)n0 * Dc, Dc, Dc / 32, btile, tid, lane);
#pragma unroll
  for (int nt = 0; nt < 4; nt++)
#pragma unroll
    for (int v = 0; v < 8; v++) {
      int n = n0 + nt * 16 + col;
      int t = m0 + hi * 8 + v;
      float val = acc[nt].e[v] + bq[n];
      int which = n >> 10, r = n & (Dc - 1);
      int h = r >> 6, hd = r & 63;
      int bb = t >> 11, s = t & (Sc - 1);
      __bf16* dst = (which == 0) ? qb : ((which == 1) ? kb : vb);
      if (which == 0) val *= 0.125f;  // 1/sqrt(64)
      dst[((size_t)(bb * Hc + h) * Sc + s) * HDc + hd] = (__bf16)val;
    }
}

// ---------------------------------------------------------------------------
// Flash attention: 4 waves/block, 16 q rows each; K/V 64x64 tiles streamed
// through LDS with double-buffered async copies. grid = (S/64, B*H).
__global__ void __launch_bounds__(128)
k_attn(const __bf16* __restrict__ qg, const __bf16* __restrict__ kg,
       const __bf16* __restrict__ vg, __bf16* __restrict__ attnb) {
  constexpr int PLD = 72;
  __shared__ __bf16 ksh[2][64 * 64];
  __shared__ __bf16 vsh[2][64 * 64];
  __shared__ __bf16 pl[4][16][PLD];
  int tid = threadIdx.x, lane = tid & 31, w = tid >> 5;
  int col = lane & 15, hi = lane >> 4;
  int bh = blockIdx.y;
  int q0 = blockIdx.x * 64 + w * 16;
  const __bf16* qp = qg + (size_t)bh * Sc * HDc;
  const __bf16* kp = kg + (size_t)bh * Sc * HDc;
  const __bf16* vp = vg + (size_t)bh * Sc * HDc;

  AF qf[2];
  loadA(qf[0], qp + (size_t)q0 * HDc, HDc, 0, lane);
  loadA(qf[1], qp + (size_t)q0 * HDc, HDc, 32, lane);

  CF o[4];
#pragma unroll
  for (int i = 0; i < 4; i++) zeroC(o[i]);
  float mrow[8], lrow[8];
#pragma unroll
  for (int v = 0; v < 8; v++) { mrow[v] = -__builtin_inff(); lrow[v] = 0.f; }

  uint32_t kls[2] = { lds_off(ksh[0]), lds_off(ksh[1]) };
  uint32_t vls[2] = { lds_off(vsh[0]), lds_off(vsh[1]) };
  issue_tile64(kp, kls[0], tid);
  issue_tile64(vp, vls[0], tid);

  constexpr int NIT = Sc / 64;
  for (int it = 0; it < NIT; it++) {
    int cur = it & 1;
    if (it + 1 < NIT) {
      issue_tile64(kp + (size_t)(it + 1) * 64 * HDc, kls[cur ^ 1], tid);
      issue_tile64(vp + (size_t)(it + 1) * 64 * HDc, vls[cur ^ 1], tid);
      WAITA(8);
    } else {
      WAITA(0);
    }
    __syncthreads();
    const __bf16* kL = ksh[cur];
    const __bf16* vL = vsh[cur];

    CF sc[4];
#pragma unroll
    for (int i = 0; i < 4; i++) zeroC(sc[i]);
    // batched fragment loads per K-half, then WMMA burst
#pragma unroll
    for (int kk = 0; kk < 2; kk++) {
      AF b0, b1, b2, b3;
      loadB_rm(b0, kL + 0 * 16 * 64, 64, kk * 32, lane);
      loadB_rm(b1, kL + 1 * 16 * 64, 64, kk * 32, lane);
      loadB_rm(b2, kL + 2 * 16 * 64, 64, kk * 32, lane);
      loadB_rm(b3, kL + 3 * 16 * 64, 64, kk * 32, lane);
      mma(sc[0], qf[kk], b0);
      mma(sc[1], qf[kk], b1);
      mma(sc[2], qf[kk], b2);
      mma(sc[3], qf[kk], b3);
    }
    // online softmax over 64 columns (rows live in lane-groups of 16)
#pragma unroll
    for (int v = 0; v < 8; v++) {
      float pm = sc[0].e[v];
#pragma unroll
      for (int nt = 1; nt < 4; nt++) pm = fmaxf(pm, sc[nt].e[v]);
#pragma unroll
      for (int off = 1; off < 16; off <<= 1) pm = fmaxf(pm, __shfl_xor(pm, off, 32));
      float mn = fmaxf(mrow[v], pm);
      float corr = expf(mrow[v] - mn);
      mrow[v] = mn;
      float ps = 0.f;
#pragma unroll
      for (int nt = 0; nt < 4; nt++) {
        sc[nt].e[v] = expf(sc[nt].e[v] - mn);
        ps += sc[nt].e[v];
      }
#pragma unroll
      for (int off = 1; off < 16; off <<= 1) ps += __shfl_xor(ps, off, 32);
      lrow[v] = lrow[v] * corr + ps;
#pragma unroll
      for (int nt = 0; nt < 4; nt++) o[nt].e[v] *= corr;
    }
    // P (C layout) -> LDS -> A layout (same-wave DS ops are in-order)
#pragma unroll
    for (int nt = 0; nt < 4; nt++)
#pragma unroll
      for (int v = 0; v < 8; v++)
        pl[w][hi * 8 + v][nt * 16 + col] = (__bf16)sc[nt].e[v];
#pragma unroll
    for (int kk = 0; kk < 2; kk++) {
      AF pf; loadA(pf, &pl[w][0][0], PLD, kk * 32, lane);
      AF b0, b1, b2, b3;
      loadB_cm(b0, vL + 0 * 16, 64, kk * 32, lane);
      loadB_cm(b1, vL + 1 * 16, 64, kk * 32, lane);
      loadB_cm(b2, vL + 2 * 16, 64, kk * 32, lane);
      loadB_cm(b3, vL + 3 * 16, 64, kk * 32, lane);
      mma(o[0], pf, b0);
      mma(o[1], pf, b1);
      mma(o[2], pf, b2);
      mma(o[3], pf, b3);
    }
    __syncthreads();
  }
  int bb = bh >> 4, h = bh & 15;
#pragma unroll
  for (int nt = 0; nt < 4; nt++)
#pragma unroll
    for (int v = 0; v < 8; v++) {
      int srow = q0 + hi * 8 + v;
      float val = o[nt].e[v] / lrow[v];
      attnb[((size_t)(bb * Sc + srow)) * Dc + h * HDc + nt * 16 + col] = (__bf16)val;
    }
}

// ---------------------------------------------------------------------------
// Out projection: y1 = attn @ wo^T + bo (f32 out).
__global__ void __launch_bounds__(128)
k_outproj(const __bf16* __restrict__ ab, const __bf16* __restrict__ wo,
          const float* __restrict__ bo, float* __restrict__ y1) {
  __shared__ __bf16 btile[2 * 64 * 32];
  int tid = threadIdx.x, lane = tid & 31, w = tid >> 5;
  int m0 = blockIdx.y * 64 + w * 16;
  int n0 = blockIdx.x * 64;
  int col = lane & 15, hi = lane >> 4;
  CF acc[4];
#pragma unroll
  for (int i = 0; i < 4; i++) zeroC(acc[i]);
  const __bf16* arow = ab + (size_t)(m0 + col) * Dc;
  gemm_kloop(acc, arow, wo + (size_t)n0 * Dc, Dc, Dc / 32, btile, tid, lane);
#pragma unroll
  for (int nt = 0; nt < 4; nt++)
#pragma unroll
    for (int v = 0; v < 8; v++) {
      int n = n0 + nt * 16 + col;
      int t = m0 + hi * 8 + v;
      y1[(size_t)t * Dc + n] = acc[nt].e[v] + bo[n];
    }
}

// ---------------------------------------------------------------------------
// LayerNorm(a + r) -> outf (f32), optional bf16 mirror.
__global__ void __launch_bounds__(256)
k_ln(const float* __restrict__ a, const float* __restrict__ r,
     const float* __restrict__ g, const float* __restrict__ be,
     float* __restrict__ outf, __bf16* __restrict__ outb) {
  __shared__ float sred[8];
  __shared__ float sbc[2];
  int lane = threadIdx.x & 31, w = threadIdx.x >> 5;
  size_t base = (size_t)blockIdx.x * Dc;
  float vals[4];
#pragma unroll
  for (int j = 0; j < 4; j++) {
    int c = threadIdx.x + j * 256;
    vals[j] = a[base + c] + r[base + c];
  }
  float s = vals[0] + vals[1] + vals[2] + vals[3];
#pragma unroll
  for (int off = 16; off > 0; off >>= 1) s += __shfl_xor(s, off, 32);
  if (lane == 0) sred[w] = s;
  __syncthreads();
  if (threadIdx.x == 0) {
    float tot = 0.f;
    for (int i = 0; i < 8; i++) tot += sred[i];
    sbc[0] = tot * (1.f / Dc);
  }
  __syncthreads();
  float mu = sbc[0];
  float vsum = 0.f;
#pragma unroll
  for (int j = 0; j < 4; j++) { float d = vals[j] - mu; vsum += d * d; }
#pragma unroll
  for (int off = 16; off > 0; off >>= 1) vsum += __shfl_xor(vsum, off, 32);
  if (lane == 0) sred[w] = vsum;
  __syncthreads();
  if (threadIdx.x == 0) {
    float tot = 0.f;
    for (int i = 0; i < 8; i++) tot += sred[i];
    sbc[1] = rsqrtf(tot * (1.f / Dc) + 1e-5f);
  }
  __syncthreads();
  float rinv = sbc[1];
#pragma unroll
  for (int j = 0; j < 4; j++) {
    int c = threadIdx.x + j * 256;
    float ov = (vals[j] - mu) * rinv * g[c] + be[c];
    outf[base + c] = ov;
    if (outb) outb[base + c] = (__bf16)ov;
  }
}

// ---------------------------------------------------------------------------
// Gate: one wave per token, argmax over 8 expert logits.
__global__ void __launch_bounds__(256)
k_gate(const float* __restrict__ x1, const float* __restrict__ gw,
       const float* __restrict__ gb, int* __restrict__ idx) {
  int lane = threadIdx.x & 31, w = threadIdx.x >> 5;
  int t = blockIdx.x * 8 + w;
  float acc[Ec] = {};
  for (int d = lane; d < Dc; d += 32) {
    float xv = x1[(size_t)t * Dc + d];
#pragma unroll
    for (int e = 0; e < Ec; e++) acc[e] += xv * gw[(size_t)e * Dc + d];
  }
#pragma unroll
  for (int e = 0; e < Ec; e++)
#pragma unroll
    for (int off = 16; off > 0; off >>= 1) acc[e] += __shfl_xor(acc[e], off, 32);
  if (lane == 0) {
    int best = 0; float bv = acc[0] + gb[0];
#pragma unroll
    for (int e = 1; e < Ec; e++) {
      float v = acc[e] + gb[e];
      if (v > bv) { bv = v; best = e; }
    }
    idx[t] = best;
  }
}

__global__ void k_zero_counts(int* counts, int* cursor) {
  if (threadIdx.x < Ec) { counts[threadIdx.x] = 0; cursor[threadIdx.x] = 0; }
}
__global__ void k_count(const int* __restrict__ idx, int* __restrict__ counts) {
  int t = blockIdx.x * blockDim.x + threadIdx.x;
  if (t < Tc) atomicAdd(&counts[idx[t]], 1);
}
__global__ void k_scan(const int* __restrict__ counts, int* __restrict__ offs) {
  if (threadIdx.x == 0) {
    int s = 0;
    for (int e = 0; e < Ec; e++) { offs[e] = s; s += counts[e]; }
  }
}
__global__ void k_scatter(const int* __restrict__ idx, const int* __restrict__ offs,
                          int* __restrict__ cursor, int* __restrict__ perm) {
  int t = blockIdx.x * blockDim.x + threadIdx.x;
  if (t < Tc) {
    int e = idx[t];
    int p = atomicAdd(&cursor[e], 1);
    perm[offs[e] + p] = t;
  }
}

// ---------------------------------------------------------------------------
// MoE GEMM1: h = gelu(x1[perm] @ w1[e]^T + b1[e]) -> hbuf bf16 (permuted rows).
__global__ void __launch_bounds__(128)
k_moe1(const __bf16* __restrict__ x1b, const __bf16* __restrict__ w1b,
       const float* __restrict__ b1, const int* __restrict__ counts,
       const int* __restrict__ offs, const int* __restrict__ perm,
       __bf16* __restrict__ hbuf) {
  __shared__ __bf16 btile[2 * 64 * 32];
  int e = blockIdx.y >> 6, mt = blockIdx.y & 63;
  int cnt = counts[e];
  if (mt * 64 >= cnt) return;       // whole block exits together
  int tid = threadIdx.x, lane = tid & 31, w = tid >> 5;
  int col = lane & 15, hi = lane >> 4;
  int off = offs[e];
  int n0 = blockIdx.x * 64;
  int lm = mt * 64 + w * 16 + col;
  int clm = lm < cnt ? lm : cnt - 1;
  int tok = perm[off + clm];
  const __bf16* arow = x1b + (size_t)tok * Dc;
  const __bf16* wb = w1b + (size_t)e * FFc * Dc + (size_t)n0 * Dc;
  CF acc[4];
#pragma unroll
  for (int i = 0; i < 4; i++) zeroC(acc[i]);
  gemm_kloop(acc, arow, wb, Dc, Dc / 32, btile, tid, lane);
#pragma unroll
  for (int nt = 0; nt < 4; nt++)
#pragma unroll
    for (int v = 0; v < 8; v++) {
      int lrow = mt * 64 + w * 16 + hi * 8 + v;
      if (lrow < cnt) {
        int n = n0 + nt * 16 + col;
        float val = acc[nt].e[v] + b1[e * FFc + n];
        val = val * 0.5f * (1.f + erff(val * 0.70710678f));  // exact GELU
        hbuf[(size_t)(off + lrow) * FFc + n] = (__bf16)val;
      }
    }
}

// MoE GEMM2: y = h @ w2[e]^T + b2[e], scatter rows back to moe[tok, :] (f32).
__global__ void __launch_bounds__(128)
k_moe2(const __bf16* __restrict__ hbuf, const __bf16* __restrict__ w2b,
       const float* __restrict__ b2, const int* __restrict__ counts,
       const int* __restrict__ offs, const int* __restrict__ perm,
       float* __restrict__ moe) {
  __shared__ __bf16 btile[2 * 64 * 32];
  int e = blockIdx.y >> 6, mt = blockIdx.y & 63;
  int cnt = counts[e];
  if (mt * 64 >= cnt) return;
  int tid = threadIdx.x, lane = tid & 31, w = tid >> 5;
  int col = lane & 15, hi = lane >> 4;
  int off = offs[e];
  int n0 = blockIdx.x * 64;
  int lm = mt * 64 + w * 16 + col;
  int clm = lm < cnt ? lm : cnt - 1;
  const __bf16* arow = hbuf + (size_t)(off + clm) * FFc;
  const __bf16* wb = w2b + (size_t)e * Dc * FFc + (size_t)n0 * FFc;
  CF acc[4];
#pragma unroll
  for (int i = 0; i < 4; i++) zeroC(acc[i]);
  gemm_kloop(acc, arow, wb, FFc, FFc / 32, btile, tid, lane);
#pragma unroll
  for (int nt = 0; nt < 4; nt++)
#pragma unroll
    for (int v = 0; v < 8; v++) {
      int lrow = mt * 64 + w * 16 + hi * 8 + v;
      if (lrow < cnt) {
        int tok = perm[off + lrow];
        int n = n0 + nt * 16 + col;
        moe[(size_t)tok * Dc + n] = acc[nt].e[v] + b2[e * Dc + n];
      }
    }
}

// ---------------------------------------------------------------------------
extern "C" void kernel_launch(void* const* d_in, const int* in_sizes, int n_in,
                              void* d_out, int out_size, void* d_ws, size_t ws_size,
                              hipStream_t stream) {
  (void)in_sizes; (void)n_in; (void)out_size; (void)ws_size;
  const float* x    = (const float*)d_in[0];
  const float* wqkv = (const float*)d_in[1];
  const float* bqkv = (const float*)d_in[2];
  const float* wo   = (const float*)d_in[3];
  const float* bo   = (const float*)d_in[4];
  const float* ln1g = (const float*)d_in[5];
  const float* ln1b = (const float*)d_in[6];
  const float* ln2g = (const float*)d_in[7];
  const float* ln2b = (const float*)d_in[8];
  const float* gw   = (const float*)d_in[9];
  const float* gb   = (const float*)d_in[10];
  const float* w1   = (const float*)d_in[11];
  const float* b1   = (const float*)d_in[12];
  const float* w2   = (const float*)d_in[13];
  const float* b2   = (const float*)d_in[14];
  float* out = (float*)d_out;

  char* p = (char*)d_ws;
  auto alloc = [&](size_t bytes) -> void* {
    void* r = (void*)p;
    p += (bytes + 255) & ~(size_t)255;
    return r;
  };
  __bf16* xb    = (__bf16*)alloc((size_t)Tc * Dc * 2);
  __bf16* wqkvb = (__bf16*)alloc((size_t)D3c * Dc * 2);
  __bf16* wob   = (__bf16*)alloc((size_t)Dc * Dc * 2);
  __bf16* w1b   = (__bf16*)alloc((size_t)Ec * FFc * Dc * 2);
  __bf16* w2b   = (__bf16*)alloc((size_t)Ec * Dc * FFc * 2);
  __bf16* qb    = (__bf16*)alloc((size_t)Tc * Dc * 2);
  __bf16* kb    = (__bf16*)alloc((size_t)Tc * Dc * 2);
  __bf16* vb    = (__bf16*)alloc((size_t)Tc * Dc * 2);
  __bf16* attnb = (__bf16*)alloc((size_t)Tc * Dc * 2);
  float*  y1    = (float*)alloc((size_t)Tc * Dc * 4);
  float*  x1    = (float*)alloc((size_t)Tc * Dc * 4);
  __bf16* x1b   = (__bf16*)alloc((size_t)Tc * Dc * 2);
  float*  moe   = (float*)alloc((size_t)Tc * Dc * 4);
  __bf16* hbuf  = (__bf16*)alloc((size_t)Tc * FFc * 2);
  int* idx    = (int*)alloc(Tc * 4);
  int* perm   = (int*)alloc(Tc * 4);
  int* counts = (int*)alloc(Ec * 4);
  int* offs   = (int*)alloc(Ec * 4);
  int* cursor = (int*)alloc(Ec * 4);

  // 1. bf16 conversions
  k_cvt_bf16<<<2048, 256, 0, stream>>>(x,    xb,    (size_t)Tc * Dc);
  k_cvt_bf16<<<2048, 256, 0, stream>>>(wqkv, wqkvb, (size_t)D3c * Dc);
  k_cvt_bf16<<<2048, 256, 0, stream>>>(wo,   wob,   (size_t)Dc * Dc);
  k_cvt_bf16<<<4096, 256, 0, stream>>>(w1,   w1b,   (size_t)Ec * FFc * Dc);
  k_cvt_bf16<<<4096, 256, 0, stream>>>(w2,   w2b,   (size_t)Ec * Dc * FFc);

  // 2. QKV projection
  k_qkv<<<dim3(D3c / 64, Tc / 64), 128, 0, stream>>>(xb, wqkvb, bqkv, qb, kb, vb);
  // 3. attention
  k_attn<<<dim3(Sc / 64, Bc * Hc), 128, 0, stream>>>(qb, kb, vb, attnb);
  // 4. out projection
  k_outproj<<<dim3(Dc / 64, Tc / 64), 128, 0, stream>>>(attnb, wob, bo, y1);
  // 5. LN1(x + y1)
  k_ln<<<Tc, 256, 0, stream>>>(x, y1, ln1g, ln1b, x1, x1b);
  // 6. routing
  k_gate<<<Tc / 8, 256, 0, stream>>>(x1, gw, gb, idx);
  k_zero_counts<<<1, 32, 0, stream>>>(counts, cursor);
  k_count<<<Tc / 256, 256, 0, stream>>>(idx, counts);
  k_scan<<<1, 32, 0, stream>>>(counts, offs);
  k_scatter<<<Tc / 256, 256, 0, stream>>>(idx, offs, cursor, perm);
  // 7-8. expert GEMMs (worst-case grid; device-side early exit on counts)
  k_moe1<<<dim3(FFc / 64, Ec * (Tc / 64)), 128, 0, stream>>>(x1b, w1b, b1, counts,
                                                             offs, perm, hbuf);
  k_moe2<<<dim3(Dc / 64, Ec * (Tc / 64)), 128, 0, stream>>>(hbuf, w2b, b2, counts,
                                                            offs, perm, moe);
  // 9. LN2(x1 + moe) -> output
  k_ln<<<Tc, 256, 0, stream>>>(x1, moe, ln2g, ln2b, out, ((__bf16*)nullptr));
}
